// DnlsLoss_62921270887103
// MI455X (gfx1250) — compile-verified, matrix operand order checked by reading, not run on hardware
//
#include <hip/hip_runtime.h>
#include <hip/hip_fp16.h>
#include <math.h>

typedef __attribute__((ext_vector_type(16))) _Float16     v16h;
typedef __attribute__((ext_vector_type(8)))  float        v8f;
typedef __attribute__((ext_vector_type(2)))  float        v2f;
typedef __attribute__((ext_vector_type(4)))  unsigned int v4u;
typedef __attribute__((ext_vector_type(8)))  int          v8i;
typedef __attribute__((ext_vector_type(4)))  int          v4i;

// Problem constants (B=1, T=5, C=3, H=W=256, ws=9, wt=1, ps=7, k=10, stride0=4)
constexpr int kT   = 5;
constexpr int kC   = 3;
constexpr int kH   = 256;
constexpr int kW   = 256;
constexpr int kS0  = 4;
constexpr int kNH  = 64;
constexpr int kNW  = 64;
constexpr int kKD  = 147;    // 7*7*3 patch dim
constexpr int kKP  = 160;    // padded to multiple of 32
constexpr int kNS  = 243;    // (2*wt+1)*ws*ws candidate offsets
constexpr int kSELF = 121;   // wt*ws*ws + (ws/2)*ws + ws/2
constexpr float kEPS = 1e-3f;
constexpr int kNTOT = kT * kNH * kNW * 9;  // 184320 loss terms

// LDS window: 15 rows x 75 cols per (frame,channel) plane; 9 planes
constexpr int kWR = 15;
constexpr int kWC = 75;
constexpr int kPL = kWR * kWC;            // 1125 elements per plane
constexpr int kWTOT = 9 * kPL;            // 10125 elements total
// Aliased buffer carve (bytes): [0, 40512) f32 staging; after conversion:
//   [0, 20480)      f16 window (needs 20250 B)
//   [20480, 36032)  f32 dist matrix 16x243 (needs 15552 B)
constexpr int kDistOff = 20480;

__device__ __forceinline__ int iclip(int v, int lo, int hi) {
  return v < lo ? lo : (v > hi ? hi : v);
}
__device__ __forceinline__ float img(const float* __restrict__ p, int t, int ch, int r, int c) {
  return p[((t * kC + ch) * kH + r) * kW + c];
}

__global__ void zero_out(float* o) { o[0] = 0.0f; }

// ---- search pass: f16 WMMA ranking distances, LDS-window or global-gather B ----
template <bool UL>
__device__ __forceinline__ void search_pass(
    const float* __restrict__ noisy, const _Float16* sW16, const _Float16* sQ16,
    const float* sQN, const int (*sDX)[16], const int (*sDY)[16], float* sDist,
    int t, int h, int tw, int wave, int hlf, int nidx, bool have, int tgt,
    int R0c, int C0c) {
  for (int s = wave; s < kNS; s += 4) {
    int ti = s / 81, hi = (s / 9) % 9, wi = s % 9;
    int dt = ti - 1, dh = hi - 4, dw = wi - 4;
    int ct = iclip(t + dt, 0, kT - 1);
    int cr = iclip(kS0 * h + dh + sDY[ti][nidx], 0, kH - 1);
    int cc = iclip(kS0 * (tw * 16 + nidx) + dw + sDX[ti][nidx], 0, kW - 1);
    int crb = cr - R0c, ccb = cc - C0c;
    const _Float16* plane = &sW16[ti * 3 * kPL];
    v8f acc = {};
    float cn = 0.0f;
#pragma unroll
    for (int kk = 0; kk < 5; ++kk) {
      v16h am, bm;
#pragma unroll
      for (int v = 0; v < 16; ++v) {
        int g = v >> 1, sb = v & 1;
        int k = kk * 32 + ((g < 4) ? 0 : 16) + 8 * hlf + 2 * (g & 3) + sb;
        _Float16 cv = (_Float16)0.0f;
        if (k < kKD) {
          int i = k / 21, j = (k / 3) % 7, ch = k % 3;
          if (UL) {
            int rw = iclip(crb + i - 3, 0, kWR - 1);
            int cw = iclip(ccb + j - 3, 0, kWC - 1);
            cv = plane[(ch * kWR + rw) * kWC + cw];
          } else {
            cv = (_Float16)img(noisy, ct, ch, iclip(cr + i - 3, 0, kH - 1),
                                              iclip(cc + j - 3, 0, kW - 1));
          }
        }
        float cf = (float)cv;
        cn += cf * cf;
        bm[v] = cv;
        am[v] = sQ16[nidx * kKP + k];
      }
      acc = __builtin_amdgcn_wmma_f32_16x16x32_f16(
          false, am, false, bm, (short)0, acc, false, false);
    }
    cn += __shfl_xor(cn, 16, 32);
    float dot = 0.0f;
#pragma unroll
    for (int r = 0; r < 8; ++r) dot = (tgt == r) ? acc[r] : dot;
    if (have) sDist[nidx * kNS + s] = sQN[nidx] + cn - 2.0f * dot;
  }
}

// One workgroup = 16 queries (one W-segment of a query row). 4 waves (wave32).
__global__ __launch_bounds__(128) void dnls_loss_kernel(
    const float* __restrict__ noisy, const float* __restrict__ deno,
    const float* __restrict__ fflow, const float* __restrict__ bflow,
    float* __restrict__ out) {
  __shared__ __align__(16) char sBuf[40512];  // f32 window -> {f16 window | dists}
  __shared__ _Float16 sQ16[16 * kKP];
  __shared__ float    sQD[16 * kKP];
  __shared__ float    sQN[16], sQDN[16];
  __shared__ int      sDX[3][16], sDY[3][16];
  __shared__ int      sIdx[16 * 9];
  __shared__ float    sRed[4];
  __shared__ int      sFlowNZ;

  float*     sWF   = (float*)sBuf;                    // staging (phase 1)
  _Float16*  sW16  = (_Float16*)sBuf;                 // f16 window (phase 2+)
  float*     sDist = (float*)(sBuf + kDistOff);       // distances (phase 2+)

  const int bid = blockIdx.x;
  const int tw  = bid & 3;
  const int h   = (bid >> 2) & 63;
  const int t   = bid >> 8;
  const int tid = threadIdx.x;
  const int wave = tid >> 5, lane = tid & 31;
  const int hlf = lane >> 4, nidx = lane & 15;
  const int R0c = iclip(kS0 * h - 7, 0, kH - kWR);
  const int C0c = iclip(64 * tw - 7, 0, kW - kWC);

  __builtin_prefetch(&noisy[((t * kC) * kH + kS0 * h) * kW], 0, 1);

  if (tid < 16) { sQN[tid] = 0.0f; sQDN[tid] = 0.0f; }
  if (tid == 0) sFlowNZ = 0;
  __syncthreads();

  // ---- stage query patches (noisy->f16, deno->f32) + norms ----
  for (int e = tid; e < 16 * kKP; e += 128) {
    int m = e / kKP, k = e - m * kKP;
    float qv = 0.0f, dv = 0.0f;
    if (k < kKD) {
      int i = k / 21, j = (k / 3) % 7, ch = k % 3;
      int r = iclip(kS0 * h + i - 3, 0, kH - 1);
      int c = iclip(kS0 * (tw * 16 + m) + j - 3, 0, kW - 1);
      qv = img(noisy, t, ch, r, c);
      dv = img(deno,  t, ch, r, c);
      atomicAdd(&sQN[m],  qv * qv);
      atomicAdd(&sQDN[m], dv * dv);
    }
    sQ16[m * kKP + k] = (_Float16)qv;
    sQD [m * kKP + k] = dv;
  }
  // ---- per-query rounded flow offsets, and flow-nonzero flag ----
  if (tid < 48) {
    int ti = tid >> 4, m = tid & 15;
    int dt = ti - 1, dx = 0, dy = 0;
    if (dt != 0) {
      const float* fp = (dt > 0) ? fflow : bflow;
      int r = kS0 * h, c = kS0 * (tw * 16 + m);
      float ad = fabsf((float)dt);
      dx = __float2int_rn(fp[((t * 2 + 0) * kH + r) * kW + c] * ad);
      dy = __float2int_rn(fp[((t * 2 + 1) * kH + r) * kW + c] * ad);
      if ((dx | dy) != 0) atomicOr(&sFlowNZ, 1);
    }
    sDX[ti][m] = dx; sDY[ti][m] = dy;
  }
  __syncthreads();
  const bool useLds = (sFlowNZ == 0);  // block-uniform

  // ---- stage 3x3 (frame,channel) 15x75 f32 window into LDS, then cvt f16 ----
  if (useLds) {
#if __has_builtin(__builtin_amdgcn_tensor_load_to_lds)
    if (wave == 0) {
#pragma unroll 1
      for (int pl = 0; pl < 9; ++pl) {
        int fti = pl / 3, ch = pl - fti * 3;
        int ct = iclip(t + fti - 1, 0, kT - 1);
        const float* base = noisy + (((size_t)ct * kC + ch) * kH + R0c) * kW + C0c;
        unsigned long long ga = (unsigned long long)(uintptr_t)base;
        unsigned int ldsb = (unsigned int)(uintptr_t)&sWF[pl * kPL];
        // D# group0: count=1 | lds_addr | global_addr | type=2
        v4u g0 = { 1u, ldsb, (unsigned int)ga,
                   (unsigned int)((ga >> 32) & 0x01FFFFFFu) | 0x80000000u };
        // D# group1: data_size=4B, tensor 256x256 f32, tile 75x15, stride0=256
        v8i g1 = { (int)(2u << 16),
                   (int)((unsigned)(kW & 0xFFFF) << 16),
                   (int)(((unsigned)kW >> 16) | ((unsigned)(kH & 0xFFFF) << 16)),
                   (int)(((unsigned)kH >> 16) | ((unsigned)kWC << 16)),
                   (int)kWR,
                   (int)kW,
                   0, 0 };
        v4i gz = { 0, 0, 0, 0 };
#if defined(__clang_major__) && (__clang_major__ >= 23)
        v8i gz8 = { 0, 0, 0, 0, 0, 0, 0, 0 };
        __builtin_amdgcn_tensor_load_to_lds(g0, g1, gz, gz, gz8, 0);
#else
        __builtin_amdgcn_tensor_load_to_lds(g0, g1, gz, gz, 0);
#endif
      }
      __builtin_amdgcn_s_wait_tensorcnt(0);
    }
    __syncthreads();
    // In-place f32 -> f16 conversion, barrier-staged rounds (read-then-write)
    for (int r0 = 0; r0 < (kWTOT + 1023) / 1024; ++r0) {
      float tmp[8];
      int base = r0 * 1024;
#pragma unroll
      for (int i = 0; i < 8; ++i) {
        int e = base + tid + i * 128;
        tmp[i] = (e < kWTOT) ? sWF[e] : 0.0f;
      }
      __syncthreads();
#pragma unroll
      for (int i = 0; i < 8; ++i) {
        int e = base + tid + i * 128;
        if (e < kWTOT) sW16[e] = (_Float16)tmp[i];
      }
      __syncthreads();
    }
#else
    // no TDM builtin: manual cooperative staging straight to f16
    for (int e = tid; e < kWTOT; e += 128) {
      int pl = e / kPL, rem = e - pl * kPL;
      int fti = pl / 3, ch = pl - fti * 3;
      int ct = iclip(t + fti - 1, 0, kT - 1);
      int rw = rem / kWC, cw = rem - rw * kWC;
      sW16[e] = (_Float16)img(noisy, ct, ch, R0c + rw, C0c + cw);
    }
    __syncthreads();
#endif
  }
  __syncthreads();

  const bool have = (hlf == 0) ? (nidx < 8) : (nidx >= 8);
  const int  tgt  = (hlf == 0) ? nidx : (nidx - 8);

  // ---- search: 243 offsets, f16 WMMA (ranking only) ----
  if (useLds)
    search_pass<true>(noisy, sW16, sQ16, sQN, sDX, sDY, sDist,
                      t, h, tw, wave, hlf, nidx, have, tgt, R0c, C0c);
  else
    search_pass<false>(noisy, sW16, sQ16, sQN, sDX, sDY, sDist,
                       t, h, tw, wave, hlf, nidx, have, tgt, R0c, C0c);
  __syncthreads();

  // ---- top-9 selection per query (self excluded) ----
  if (tid < 16) {
    float bd[9]; int bi[9];
#pragma unroll
    for (int r = 0; r < 9; ++r) { bd[r] = 3.4e38f; bi[r] = 0; }
    for (int s = 0; s < kNS; ++s) {
      if (s == kSELF) continue;
      float d = sDist[tid * kNS + s];
      if (d < bd[8]) {
        int r = 8;
        while (r > 0 && bd[r - 1] > d) { bd[r] = bd[r - 1]; bi[r] = bi[r - 1]; --r; }
        bd[r] = d; bi[r] = s;
      }
    }
#pragma unroll
    for (int r = 0; r < 9; ++r) sIdx[tid * 9 + r] = bi[r];
  }
  __syncthreads();

  // ---- refine: deno-query vs noisy-neighbor distances, full-f32 WMMA ----
  float local = 0.0f;
  for (int j = wave; j < 9; j += 4) {
    int s  = sIdx[nidx * 9 + j];
    int ti = s / 81, hi = (s / 9) % 9, wi = s % 9;
    int dt = ti - 1, dh = hi - 4, dw = wi - 4;
    int ct = iclip(t + dt, 0, kT - 1);
    int cr = iclip(kS0 * h + dh + sDY[ti][nidx], 0, kH - 1);
    int cc = iclip(kS0 * (tw * 16 + nidx) + dw + sDX[ti][nidx], 0, kW - 1);
    v8f acc = {};
    float cn = 0.0f;
#pragma unroll 4
    for (int kk = 0; kk < 40; ++kk) {
      int k0 = 4 * kk + 2 * hlf;
      v2f am, bm;
#pragma unroll
      for (int e = 0; e < 2; ++e) {
        int k = k0 + e;
        float cv = 0.0f;
        if (k < kKD) {
          int i = k / 21, jj = (k / 3) % 7, ch = k % 3;
          cv = img(noisy, ct, ch, iclip(cr + i - 3, 0, kH - 1),
                                  iclip(cc + jj - 3, 0, kW - 1));
        }
        cn += cv * cv;
        bm[e] = cv;
        am[e] = sQD[nidx * kKP + k];
      }
      acc = __builtin_amdgcn_wmma_f32_16x16x4_f32(
          false, am, false, bm, (short)0, acc, false, false);
    }
    cn += __shfl_xor(cn, 16, 32);
    float dot = 0.0f;
#pragma unroll
    for (int r = 0; r < 8; ++r) dot = (tgt == r) ? acc[r] : dot;
    if (have) local += sqrtf(sQDN[nidx] + cn - 2.0f * dot + kEPS);
  }

  // ---- reduction: wave -> block -> global atomic (mean) ----
#pragma unroll
  for (int off = 16; off > 0; off >>= 1) local += __shfl_xor(local, off, 32);
  if (lane == 0) sRed[wave] = local;
  __syncthreads();
  if (tid == 0) {
    float bs = sRed[0] + sRed[1] + sRed[2] + sRed[3];
    atomicAdd(out, bs * (1.0f / (float)kNTOT));
  }
}

extern "C" void kernel_launch(void* const* d_in, const int* in_sizes, int n_in,
                              void* d_out, int out_size, void* d_ws, size_t ws_size,
                              hipStream_t stream) {
  (void)in_sizes; (void)n_in; (void)out_size; (void)d_ws; (void)ws_size;
  const float* noisy = (const float*)d_in[0];
  const float* deno  = (const float*)d_in[1];
  const float* ff    = (const float*)d_in[2];
  const float* bf    = (const float*)d_in[3];
  float* out = (float*)d_out;

  zero_out<<<1, 1, 0, stream>>>(out);
  // 5 frames * 64 rows * 4 tiles-of-16-queries = 1280 workgroups, 4 waves each
  dnls_loss_kernel<<<dim3(kT * kNH * (kNW / 16)), dim3(128), 0, stream>>>(
      noisy, deno, ff, bf, out);
}
